// STGCN_82334523064538
// MI455X (gfx1250) — compile-verified, matrix operand order checked by reading, not run on hardware
//
#include <hip/hip_runtime.h>
#include <hip/hip_bf16.h>

// ---------------------------------------------------------------------------
// ST-GCN for MI455X (gfx1250), wave32. All dense contractions via
// v_wmma_f32_16x16x32_f16 with branch-free, b128-vectorized fragment loads:
//  - activations stored f16 with zero-padded channel stride Sp=max(C,32)
//  - weights repacked on-device into WMMA B-fragment order
//  - edge scatter replaced by a dense padded 80x96 adjacency GEMM whose A side
//    is pre-laid-out in WMMA A-fragment order; B fragments are register-reused
//    across all five row tiles of each (b,t) slice.
// ---------------------------------------------------------------------------

typedef __attribute__((ext_vector_type(16))) _Float16 v16h;
typedef __attribute__((ext_vector_type(8)))  _Float16 hv8;
typedef __attribute__((ext_vector_type(8)))  float    v8f;

#define Bn  32
#define Tt  64
#define Nn  68
#define Ee  128
#define Mrows 139264LL        // B*T*N
#define EPSs 1e-5f

__device__ __forceinline__ v8f wmma16(v16h a, v16h b, v8f c) {
  return __builtin_amdgcn_wmma_f32_16x16x32_f16(false, a, false, b, (short)0, c,
                                                false, false);
}

// ---------------------------------------------------------------------------
// Plain GEMM: Out[M,Sp] = A[M,Sp] @ Wf, f16 in, f32 acc, f16 out.
// Fully unpredicated: M grid-exact, K=Sp padded, O=Sp padded (Wf zero-padded).
// One wave per 16x16 tile; inner loop = 4x b128 loads + 1 wmma.
// ---------------------------------------------------------------------------
__global__ void __launch_bounds__(128)
gemm_xw_f16(const _Float16* __restrict__ A, const _Float16* __restrict__ Wf,
            _Float16* __restrict__ Out, int SpIn, int SpOut, int otiles) {
  const int lane = threadIdx.x & 31;
  const int wave = threadIdx.x >> 5;
  const int m0 = (blockIdx.x * 4 + wave) * 16;
  const int otile = blockIdx.y;
  const int kbl = (lane >> 4) << 3;
  const _Float16* rowA = A + (size_t)(m0 + (lane & 15)) * SpIn + kbl;
  const _Float16* wf = Wf + ((size_t)otile * 32 + lane) * 16;
  const int ktiles = SpIn >> 5;
  v8f acc = {};
  for (int kt = 0; kt < ktiles; ++kt) {
    __builtin_prefetch(rowA + kt * 32 + 32, 0, 1);          // global_prefetch_b8
    hv8 a0 = *(const hv8*)(rowA + kt * 32);
    hv8 a1 = *(const hv8*)(rowA + kt * 32 + 16);
    const _Float16* wp = wf + (size_t)kt * otiles * 512;
    hv8 b0 = *(const hv8*)(wp);
    hv8 b1 = *(const hv8*)(wp + 8);
    v16h a, b;
#pragma unroll
    for (int e = 0; e < 8; ++e) {
      a[e] = a0[e]; a[e + 8] = a1[e];
      b[e] = b0[e]; b[e + 8] = b1[e];
    }
    acc = wmma16(a, b, acc);
  }
  _Float16* outp = Out + (size_t)(m0 + ((lane >> 4) << 3)) * SpOut
                       + otile * 16 + (lane & 15);
#pragma unroll
  for (int r = 0; r < 8; ++r) outp[(size_t)r * SpOut] = (_Float16)acc[r];
}

// ---------------------------------------------------------------------------
// Batched aggregation: per (b,t) slice Agg[68,Sp] = Anorm[68,68] @ H[68,Sp].
// One wave per (bt, otile): the three 32x16 B fragments are loaded ONCE into
// registers and reused across all 5 row tiles (15 WMMAs per wave). A side is
// pre-packed adjacency fragments -> two b128 loads per tile.
// ---------------------------------------------------------------------------
__global__ void __launch_bounds__(128)
gemm_agg_f16(const _Float16* __restrict__ Anf, const _Float16* __restrict__ H,
             _Float16* __restrict__ Agg, int Sp) {
  const int lane = threadIdx.x & 31;
  const int wave = threadIdx.x >> 5;
  const int bt = blockIdx.x * 4 + wave;
  const int otile = blockIdx.y;
  const int kbl = (lane >> 4) << 3;
  const int oB = otile * 16 + (lane & 15);
  const _Float16* Hs = H + (size_t)bt * Nn * Sp + oB;

  // Preload B fragments (k = 0..95; rows >= 68 are neighbors/pad multiplied by
  // exact zeros in the padded adjacency).
  v16h bfr[3];
#pragma unroll
  for (int kt = 0; kt < 3; ++kt) {
#pragma unroll
    for (int e = 0; e < 16; ++e) {
      int k = kt * 32 + kbl + (e < 8 ? e : e + 8);
      bfr[kt][e] = Hs[(size_t)k * Sp];
    }
  }

#pragma unroll
  for (int ntile = 0; ntile < 5; ++ntile) {
    v8f acc = {};
#pragma unroll
    for (int kt = 0; kt < 3; ++kt) {
      const _Float16* ap = Anf + ((size_t)(kt * 5 + ntile) * 32 + lane) * 16;
      hv8 a0 = *(const hv8*)ap;
      hv8 a1 = *(const hv8*)(ap + 8);
      v16h a;
#pragma unroll
      for (int e = 0; e < 8; ++e) { a[e] = a0[e]; a[e + 8] = a1[e]; }
      acc = wmma16(a, bfr[kt], acc);
    }
    const int nr = ntile * 16 + ((lane >> 4) << 3);
#pragma unroll
    for (int r = 0; r < 8; ++r) {
      int n = nr + r;
      if (n < Nn) Agg[((size_t)bt * Nn + n) * Sp + oB] = (_Float16)acc[r];
    }
  }
}

// ---------------------------------------------------------------------------
// Implicit-GEMM conv over the node axis: K = Cin*kk. For Cin>=8 each 8-wide
// per-lane chunk stays inside one tap -> single b128 gather per chunk.
// ---------------------------------------------------------------------------
__global__ void __launch_bounds__(128)
gemm_tcn_f16(const _Float16* __restrict__ Hbn, const _Float16* __restrict__ Wf,
             _Float16* __restrict__ Y, int Cin, int cshift, int kk, int Sp,
             int otiles) {
  const int lane = threadIdx.x & 31;
  const int wave = threadIdx.x >> 5;
  const int m0 = (blockIdx.x * 4 + wave) * 16;
  const int otile = blockIdx.y;
  const int K = Cin * kk;
  const int ktiles = (K + 31) >> 5;
  const int pad = (kk - 1) >> 1;
  const int m = m0 + (lane & 15);
  const int bt = m / Nn;
  const int n = m - bt * Nn;
  const int kbl = (lane >> 4) << 3;
  const _Float16* Hs = Hbn + (size_t)bt * Nn * Sp;
  const _Float16* wf = Wf + ((size_t)otile * 32 + lane) * 16;
  const hv8 zero8 = {};
  v8f acc = {};
  for (int kt = 0; kt < ktiles; ++kt) {
    const _Float16* wp = wf + (size_t)kt * otiles * 512;
    hv8 b0 = *(const hv8*)(wp);
    hv8 b1 = *(const hv8*)(wp + 8);
    v16h a, b;
#pragma unroll
    for (int e = 0; e < 8; ++e) { b[e] = b0[e]; b[e + 8] = b1[e]; }
    if (Cin >= 8) {
#pragma unroll
      for (int h = 0; h < 2; ++h) {
        int kq0 = kt * 32 + kbl + h * 16;
        hv8 av = zero8;
        if (kq0 < K) {
          int d  = kq0 >> cshift;
          int c0 = kq0 & (Cin - 1);
          int nn = n + d - pad;
          if (nn >= 0 && nn < Nn) av = *(const hv8*)(Hs + (size_t)nn * Sp + c0);
        }
#pragma unroll
        for (int e = 0; e < 8; ++e) a[h * 8 + e] = av[e];
      }
    } else {
#pragma unroll
      for (int e = 0; e < 16; ++e) {
        int kq = kt * 32 + kbl + (e < 8 ? e : e + 8);
        _Float16 v = (_Float16)0.f;
        if (kq < K) {
          int d  = kq >> cshift;
          int c  = kq & (Cin - 1);
          int nn = n + d - pad;
          if (nn >= 0 && nn < Nn) v = Hs[(size_t)nn * Sp + c];
        }
        a[e] = v;
      }
    }
    acc = wmma16(a, b, acc);
  }
  _Float16* outp = Y + (size_t)(m0 + ((lane >> 4) << 3)) * Sp
                     + otile * 16 + (lane & 15);
#pragma unroll
  for (int r = 0; r < 8; ++r) outp[(size_t)r * Sp] = (_Float16)acc[r];
}

// ---------------------------------------------------------------------------
// Build normalized adjacency and emit it directly in WMMA A-fragment order:
// Anf[(kt*5+ntile)*512 + lane*16 + e], 80x96 zero-padded. One block, LDS.
// ---------------------------------------------------------------------------
__global__ void build_adj(const int* __restrict__ ei, const float* __restrict__ ew,
                          _Float16* __restrict__ Anf) {
  __shared__ float deg[Nn];
  __shared__ float dinv[Nn];
  __shared__ float A[80 * 96];
  const int tid = threadIdx.x;                      // 128 threads
  for (int i = tid; i < Nn; i += 128) deg[i] = 0.f;
  for (int i = tid; i < 80 * 96; i += 128) A[i] = 0.f;
  __syncthreads();
  if (tid < Ee) atomicAdd(&deg[ei[Ee + tid]], ew[tid]);               // cols
  for (int nid = tid; nid < Nn; nid += 128) atomicAdd(&deg[nid], 1.f); // loops
  __syncthreads();
  for (int nid = tid; nid < Nn; nid += 128)
    dinv[nid] = deg[nid] > 0.f ? rsqrtf(deg[nid]) : 0.f;
  __syncthreads();
  if (tid < Ee) {
    int r = ei[tid], c = ei[Ee + tid];
    atomicAdd(&A[c * 96 + r], dinv[r] * ew[tid] * dinv[c]);
  }
  for (int nid = tid; nid < Nn; nid += 128)
    atomicAdd(&A[nid * 96 + nid], dinv[nid] * dinv[nid]);
  __syncthreads();
  for (int i = tid; i < 3 * 5 * 512; i += 128) {
    int e = i & 15;
    int lane = (i >> 4) & 31;
    int tile = i >> 9;
    int ntile = tile % 5;
    int kt = tile / 5;
    int n = ntile * 16 + (lane & 15);
    int k = kt * 32 + ((lane >> 4) << 3) + (e < 8 ? e : e + 8);
    Anf[i] = (_Float16)A[n * 96 + k];
  }
}

// ---------------------------------------------------------------------------
// Weight repack into B-fragment order (zero-padded K and O).
// ---------------------------------------------------------------------------
__global__ void repack_frag_xw(const float* __restrict__ W, _Float16* __restrict__ Wf,
                               int K, int O, int otiles, long long total) {
  long long i = (long long)blockIdx.x * blockDim.x + threadIdx.x;
  if (i >= total) return;
  int e = (int)(i & 15);
  int lane = (int)((i >> 4) & 31);
  long long tile = i >> 9;
  int otile = (int)(tile % otiles);
  int kt = (int)(tile / otiles);
  int o = otile * 16 + (lane & 15);
  int k = kt * 32 + ((lane >> 4) << 3) + (e < 8 ? e : e + 8);
  float v = (k < K && o < O) ? W[(size_t)k * O + o] : 0.f;
  Wf[i] = (_Float16)v;
}

// tcn_W is [O][Cin][kk]; GEMM k-index kq -> (d = kq/Cin, c = kq%Cin).
__global__ void repack_frag_tcn(const float* __restrict__ W, _Float16* __restrict__ Wf,
                                int Cin, int kk, int O, int otiles, long long total) {
  long long i = (long long)blockIdx.x * blockDim.x + threadIdx.x;
  if (i >= total) return;
  int e = (int)(i & 15);
  int lane = (int)((i >> 4) & 31);
  long long tile = i >> 9;
  int otile = (int)(tile % otiles);
  int kt = (int)(tile / otiles);
  int o = otile * 16 + (lane & 15);
  int kq = kt * 32 + ((lane >> 4) << 3) + (e < 8 ? e : e + 8);
  const int K = Cin * kk;
  float v = 0.f;
  if (kq < K && o < O) {
    int d = kq / Cin;
    int c = kq - d * Cin;
    v = W[((size_t)o * Cin + c) * kk + d];
  }
  Wf[i] = (_Float16)v;
}

// ---------------------------------------------------------------------------
// Elementwise / reduction helpers (padded stride Sp, power of two)
// ---------------------------------------------------------------------------
__global__ void cvt_pad_f16(const float* __restrict__ in, _Float16* __restrict__ out,
                            int K, int Sp, long long total) {
  long long i = (long long)blockIdx.x * blockDim.x + threadIdx.x;
  if (i >= total) return;
  int c = (int)(i & (Sp - 1));
  long long r = i / Sp;
  out[i] = (c < K) ? (_Float16)in[r * K + c] : (_Float16)0.f;
}

__global__ void zero_f32(float* __restrict__ p, int n) {
  int i = blockIdx.x * blockDim.x + threadIdx.x;
  if (i < n) p[i] = 0.f;
}

// Per-channel sum/sumsq over rows; C power of two <= 256, row stride Sp.
__global__ void __launch_bounds__(256)
reduce_stats(const _Float16* __restrict__ X, int C, int Sp, long long rows,
             float* __restrict__ sums) {
  const int tid = threadIdx.x;
  const int lpr = 256 / C;
  const int c = tid & (C - 1);
  long long r = (long long)blockIdx.x * lpr + (tid / C);
  const long long stride = (long long)gridDim.x * lpr;
  float sm = 0.f, sq = 0.f;
  for (; r < rows; r += stride) {
    float v = (float)X[r * Sp + c];
    sm += v; sq += v * v;
  }
  atomicAdd(&sums[c], sm);
  atomicAdd(&sums[C + c], sq);
}

// hbn = relu(BN(agg)) -> f16; pad columns forced to zero.
__global__ void bn_relu_f16(const _Float16* __restrict__ X, _Float16* __restrict__ Out,
                            const float* __restrict__ sums, const float* __restrict__ g,
                            const float* __restrict__ b, int C, int Sp, long long total) {
  long long i = (long long)blockIdx.x * blockDim.x + threadIdx.x;
  if (i >= total) return;
  int c = (int)(i & (Sp - 1));
  if (c >= C) { Out[i] = (_Float16)0.f; return; }
  float mean = sums[c] * (1.f / (float)Mrows);
  float var  = sums[C + c] * (1.f / (float)Mrows) - mean * mean;
  float y = g[c] * ((float)X[i] - mean) * rsqrtf(var + EPSs) + b[c];
  Out[i] = (_Float16)fmaxf(y, 0.f);
}

// x_next = relu(BN2(y) + BN3(res)) -> f16; pad columns forced to zero.
__global__ void combine_f16(const _Float16* __restrict__ Y, const _Float16* __restrict__ R,
                            _Float16* __restrict__ Xout,
                            const float* __restrict__ sY, const float* __restrict__ sR,
                            const float* __restrict__ g2, const float* __restrict__ b2,
                            const float* __restrict__ g3, const float* __restrict__ b3,
                            int C, int Sp, long long total) {
  long long i = (long long)blockIdx.x * blockDim.x + threadIdx.x;
  if (i >= total) return;
  int c = (int)(i & (Sp - 1));
  if (c >= C) { Xout[i] = (_Float16)0.f; return; }
  float mY = sY[c] * (1.f / (float)Mrows);
  float vY = sY[C + c] * (1.f / (float)Mrows) - mY * mY;
  float mR = sR[c] * (1.f / (float)Mrows);
  float vR = sR[C + c] * (1.f / (float)Mrows) - mR * mR;
  float h  = g2[c] * ((float)Y[i] - mY) * rsqrtf(vY + EPSs) + b2[c];
  float rr = g3[c] * ((float)R[i] - mR) * rsqrtf(vR + EPSs) + b3[c];
  Xout[i] = (_Float16)fmaxf(h + rr, 0.f);
}

// z0[b,w] = sum_{t,n} x[b,t,n,w] * fcW[t,n] + fcb   (Sp == C == 256 here)
__global__ void __launch_bounds__(256)
fc_contract(const _Float16* __restrict__ X, const float* __restrict__ fcW,
            const float* __restrict__ fcb, float* __restrict__ z0) {
  const int b = blockIdx.x, w = threadIdx.x;
  const _Float16* xb = X + (size_t)b * Tt * Nn * 256;
  float acc = 0.f;
  for (int t = 0; t < Tt; ++t)
    for (int n = 0; n < Nn; ++n)
      acc += (float)xb[((size_t)t * Nn + n) * 256 + w] * fcW[t * Nn + n];
  z0[b * 256 + w] = acc + fcb[0];
}

__device__ void mlp_stage(const float* in, float* out, int din, int dout,
                          const float* W, const float* bias, bool relu,
                          int tid, int nth) {
  for (int idx = tid; idx < Bn * dout; idx += nth) {
    int bb = idx / dout, j = idx - bb * dout;
    float a = bias[j];
    const float* zi = in + bb * din;
    for (int i = 0; i < din; ++i) a += zi[i] * W[i * dout + j];
    out[bb * dout + j] = relu ? fmaxf(a, 0.f) : a;
  }
}

__global__ void __launch_bounds__(256)
classifier(const float* __restrict__ z0,
           const float* W0, const float* b0, const float* W1, const float* b1,
           const float* W2, const float* b2, const float* W3, const float* b3,
           float* __restrict__ out) {
  __shared__ float bufA[Bn * 256];
  __shared__ float bufB[Bn * 128];
  const int tid = threadIdx.x, nth = blockDim.x;
  for (int i = tid; i < Bn * 256; i += nth) bufA[i] = z0[i];
  __syncthreads();
  mlp_stage(bufA, bufB, 256, 128, W0, b0, true, tid, nth);  __syncthreads();
  mlp_stage(bufB, bufA, 128, 64,  W1, b1, true, tid, nth);  __syncthreads();
  mlp_stage(bufA, bufB, 64, 16,   W2, b2, true, tid, nth);  __syncthreads();
  mlp_stage(bufB, out, 16, 1,     W3, b3, false, tid, nth);
}

// ---------------------------------------------------------------------------
// Host orchestration
// ---------------------------------------------------------------------------
static inline int ilog2i(int v) { int s = 0; while ((1 << s) < v) ++s; return s; }

extern "C" void kernel_launch(void* const* d_in, const int* in_sizes, int n_in,
                              void* d_out, int out_size, void* d_ws, size_t ws_size,
                              hipStream_t stream) {
  (void)in_sizes; (void)n_in; (void)out_size; (void)ws_size;
  static const int CH[8] = {3, 4, 8, 16, 32, 64, 128, 256};
  static const int KS[7] = {3, 3, 3, 5, 5, 5, 5};

  const float* x_in = (const float*)d_in[0];
  const int*   ei   = (const int*)d_in[1];
  auto LP = [&](int layer, int off) -> const float* {
    return (const float*)d_in[2 + layer * 13 + off];
  };
  // offsets: 0 gcn_W, 1 gcn_b, 2 edge_w, 3 g1, 4 b1, 5 tcn_W, 6 tcn_b,
  //          7 g2, 8 b2, 9 res_W, 10 res_b, 11 g3, 12 b3
  const float* fcW = (const float*)d_in[93];
  const float* fcb = (const float*)d_in[94];

  // ---- workspace carve (f16 activations, padded channel strides) ----
  char* w = (char*)d_ws;
  const size_t actB = (size_t)Mrows * 256 * sizeof(_Float16);   // 71.3 MB
  _Float16* Xa   = (_Float16*)w;  w += actB;
  _Float16* Xb   = (_Float16*)w;  w += actB;
  _Float16* H16  = (_Float16*)w;  w += actB + 32 * 256 * sizeof(_Float16); // +pad for agg over-read
  _Float16* AGG  = (_Float16*)w;  w += actB;
  _Float16* RES  = (_Float16*)w;  w += actB;
  _Float16* Wg   = (_Float16*)w;  w += 128 * 256 * sizeof(_Float16);  // frag xw (gcn)
  _Float16* Wr   = (_Float16*)w;  w += 128 * 256 * sizeof(_Float16);  // frag xw (res)
  _Float16* Wt   = (_Float16*)w;  w += 256 * 5 * 256 * sizeof(_Float16); // frag tcn
  _Float16* Anf  = (_Float16*)w;  w += 3 * 5 * 512 * sizeof(_Float16);
  float* stats   = (float*)w;     w += 6 * 256 * sizeof(float);
  float* z0      = (float*)w;     w += Bn * 256 * sizeof(float);

  // input f32 [M,3] -> f16 padded [M,32]
  {
    long long total = Mrows * 32;
    cvt_pad_f16<<<(unsigned)((total + 255) / 256), 256, 0, stream>>>(x_in, Xa, 3, 32, total);
  }

  _Float16* xcur = Xa;
  _Float16* xnext = Xb;

  for (int li = 0; li < 7; ++li) {
    const int ci = CH[li], co = CH[li + 1], kk = KS[li];
    const int SpIn  = ci < 32 ? 32 : ci;      // padded input stride (pow2)
    const int SpOut = co < 32 ? 32 : co;      // padded output stride (pow2)
    const int otiles = SpOut / 16;
    const int ktX = SpIn / 32;
    const int ktT = (co * kk + 31) / 32;
    const long long nSp = Mrows * SpOut;
    const unsigned ewb = (unsigned)((nSp + 255) / 256);

    build_adj<<<1, 128, 0, stream>>>(ei, LP(li, 2), Anf);
    {
      long long tx = (long long)ktX * otiles * 512;
      repack_frag_xw<<<(unsigned)((tx + 255) / 256), 256, 0, stream>>>(LP(li, 0), Wg, ci, co, otiles, tx);
      repack_frag_xw<<<(unsigned)((tx + 255) / 256), 256, 0, stream>>>(LP(li, 9), Wr, ci, co, otiles, tx);
      long long tt = (long long)ktT * otiles * 512;
      repack_frag_tcn<<<(unsigned)((tt + 255) / 256), 256, 0, stream>>>(LP(li, 5), Wt, co, kk, co, otiles, tt);
    }
    zero_f32<<<6, 256, 0, stream>>>(stats, 6 * 256);

    // GCN feature transform: H = X @ gcn_W  (conv biases are BN no-ops)
    gemm_xw_f16<<<dim3(2176, otiles), 128, 0, stream>>>(xcur, Wg, H16, SpIn, SpOut, otiles);
    // Aggregation: AGG = Anorm @ H, batched over 2048 (b,t) slices;
    // one wave per (bt, otile), B fragments register-reused over 5 row tiles.
    gemm_agg_f16<<<dim3(512, otiles), 128, 0, stream>>>(Anf, H16, AGG, SpOut);
    reduce_stats<<<1024, 256, 0, stream>>>(AGG, co, SpOut, Mrows, stats + 0);

    // Residual branch: RES = X @ res_W
    gemm_xw_f16<<<dim3(2176, otiles), 128, 0, stream>>>(xcur, Wr, RES, SpIn, SpOut, otiles);
    reduce_stats<<<1024, 256, 0, stream>>>(RES, co, SpOut, Mrows, stats + 2 * 256);

    // relu(BN1(AGG)) -> H16 (TCN input)
    bn_relu_f16<<<ewb, 256, 0, stream>>>(AGG, H16, stats + 0, LP(li, 3), LP(li, 4), co, SpOut, nSp);

    // TCN implicit GEMM over node axis -> AGG (reused as Y)
    gemm_tcn_f16<<<dim3(2176, otiles), 128, 0, stream>>>(H16, Wt, AGG, co, ilog2i(co), kk, SpOut, otiles);
    reduce_stats<<<1024, 256, 0, stream>>>(AGG, co, SpOut, Mrows, stats + 4 * 256);

    // x_next = relu(BN2(Y) + BN3(RES)), keeps pad columns zeroed
    combine_f16<<<ewb, 256, 0, stream>>>(AGG, RES, xnext, stats + 4 * 256, stats + 2 * 256,
                                         LP(li, 7), LP(li, 8), LP(li, 11), LP(li, 12),
                                         co, SpOut, nSp);
    _Float16* t = xcur; xcur = xnext; xnext = t;
  }

  // head: (T,N) contraction then MLP 256->128->64->16->1
  fc_contract<<<Bn, 256, 0, stream>>>(xcur, fcW, fcb, z0);
  classifier<<<1, 256, 0, stream>>>(z0,
      (const float*)d_in[95], (const float*)d_in[96],
      (const float*)d_in[97], (const float*)d_in[98],
      (const float*)d_in[99], (const float*)d_in[100],
      (const float*)d_in[101], (const float*)d_in[102],
      (float*)d_out);
}
